// NeuroSAT_55542517072558
// MI455X (gfx1250) — compile-verified, matrix operand order checked by reading, not run on hardware
//
#include <hip/hip_runtime.h>
#include <hip/hip_bf16.h>
#include <cstdint>
#include <cstddef>

#define DIM     128
#define NVARS   5000
#define NLITS   10000
#define NCLS    20000
#define NEDGES  60000
#define NROUNDS 12
#define GDIM    512

typedef __bf16 v16bf __attribute__((ext_vector_type(16)));
typedef float  v8f   __attribute__((ext_vector_type(8)));

// ---------------------------------------------------------------- helpers

__device__ __forceinline__ unsigned short f2bf(float f) {
  uint32_t u = __float_as_uint(f);
  u += 0x7FFFu + ((u >> 16) & 1u);   // round-to-nearest-even
  return (unsigned short)(u >> 16);
}

union FragCast { uint4 q[2]; v16bf v; };

// 16-bit A/B fragment (16x32, MxK). ISA layout: lane L holds row L%16;
// halfs 0..7 = K (L>=16?8:0)+i ; halfs 8..15 = 16+(L>=16?8:0)+(i-8).
// Two contiguous 16B runs per lane -> two b128 loads.
__device__ __forceinline__ v16bf load_frag(const unsigned short* __restrict__ base,
                                           int row0, int ld, int k0, int lane) {
  int r  = row0 + (lane & 15);
  int kb = k0 + ((lane >> 4) << 3);
  const unsigned short* p = base + (size_t)r * ld + kb;
  FragCast f;
  f.q[0] = *reinterpret_cast<const uint4*>(p);
  f.q[1] = *reinterpret_cast<const uint4*>(p + 16);
  return f.v;
}

// ---------------------------------------------------------------- GEMM
// C[M,N] (f32) = act( A[M,K](bf16) * B[N,K]^T(bf16) + bias [+ C_old] )
// One wave -> one 16x64 output strip: A fragment reused across 4 B fragments,
// 4 independent accumulators (hides WMMA latency, no D->A/B hazard chain).
// Requires M % 16 == 0, N % 64 == 0, K % 32 == 0 (all shapes here qualify).
__global__ void gemm_bf16_wmma(const unsigned short* __restrict__ A, int lda,
                               const unsigned short* __restrict__ B, int ldb,
                               float* __restrict__ C, int ldc,
                               const float* __restrict__ bias,
                               int M, int N, int K, int accum, int do_relu)
{
  int wave    = blockIdx.x * (blockDim.x >> 5) + (threadIdx.x >> 5);
  int lane    = threadIdx.x & 31;
  int stripsN = N >> 6;                          // 64 columns per strip
  int strips  = (M >> 4) * stripsN;
  if (wave >= strips) return;                    // whole-wave uniform exit
  int tm = wave / stripsN, tn = wave - tm * stripsN;
  int row0 = tm << 4, col0 = tn << 6;
  int laneN  = lane & 15;
  int crow0  = row0 + ((lane >> 4) << 3);

  v8f c[4];
#pragma unroll
  for (int s = 0; s < 4; ++s) {
    int ccol = col0 + (s << 4) + laneN;
    float bv = bias ? bias[ccol] : 0.0f;
    if (accum) {
#pragma unroll
      for (int v = 0; v < 8; ++v) c[s][v] = C[(size_t)(crow0 + v) * ldc + ccol] + bv;
    } else {
#pragma unroll
      for (int v = 0; v < 8; ++v) c[s][v] = bv;
    }
  }

  for (int k0 = 0; k0 < K; k0 += 32) {
    v16bf a = load_frag(A, row0, lda, k0, lane);
#pragma unroll
    for (int s = 0; s < 4; ++s) {
      v16bf b = load_frag(B, col0 + (s << 4), ldb, k0, lane);
      c[s] = __builtin_amdgcn_wmma_f32_16x16x32_bf16(false, a, false, b,
                                                     (short)0, c[s], false, false);
    }
  }

#pragma unroll
  for (int s = 0; s < 4; ++s) {
    int ccol = col0 + (s << 4) + laneN;
#pragma unroll
    for (int v = 0; v < 8; ++v) {
      float val = c[s][v];
      if (do_relu) val = fmaxf(val, 0.0f);
      C[(size_t)(crow0 + v) * ldc + ccol] = val;
    }
  }
}

// ---------------------------------------------------------------- elementwise

__global__ void k_f32_to_bf16(const float* __restrict__ s,
                              unsigned short* __restrict__ d, int n) {
  int i = blockIdx.x * blockDim.x + threadIdx.x;
  if (i < n) d[i] = f2bf(s[i]);
}

// flipped = concat([lit_h[NVARS:], lit_h[:NVARS]]) -> bf16
__global__ void k_flip_to_bf16(const float* __restrict__ lit_h,
                               unsigned short* __restrict__ d) {
  int i = blockIdx.x * blockDim.x + threadIdx.x;
  if (i >= NLITS * DIM) return;
  int r = i >> 7, t = i & 127;
  int sr = r + NVARS; if (sr >= NLITS) sr -= NLITS;
  d[i] = f2bf(lit_h[(size_t)sr * DIM + t]);
}

__global__ void k_init_h(float* __restrict__ h, const float* __restrict__ w,
                         const float* __restrict__ b, int n) {
  int i = blockIdx.x * blockDim.x + threadIdx.x;
  if (i < n) h[i] = w[i & 127] + b[i & 127];
}

__global__ void k_zero(float* __restrict__ p, int n) {
  int i = blockIdx.x * blockDim.x + threadIdx.x;
  if (i < n) p[i] = 0.0f;
}

__device__ __forceinline__ float sigm(float x) { return 1.0f / (1.0f + __expf(-x)); }

// ---------------------------------------------------------------- edge phases
// lit -> clause: clause c owns edges 3c..3c+2 (edge_dst = repeat(arange)).
__global__ void k_edge_l2c(const int* __restrict__ esrc,
                           const float* __restrict__ Msrc,   // lit_h @ la_ws^T
                           const float* __restrict__ Mdst,   // clause_h @ la_wd^T
                           const float* __restrict__ b1,
                           const float* __restrict__ w2,
                           const float* __restrict__ b2,
                           const float* __restrict__ lmout,
                           float* __restrict__ agg)
{
  int c = blockIdx.x, t = threadIdx.x;
  __shared__ float red[DIM];
  __shared__ float attn_s;
  float mdst = Mdst[(size_t)c * DIM + t] + b1[t];
  float acc = 0.0f;
  for (int j = 0; j < 3; ++j) {
    int s = esrc[c * 3 + j];
    float e = fmaxf(Msrc[(size_t)s * DIM + t] + mdst, 0.0f);
    red[t] = e * w2[t];
    __syncthreads();
    for (int off = 64; off > 0; off >>= 1) {
      if (t < off) red[t] += red[t + off];
      __syncthreads();
    }
    if (t == 0) attn_s = sigm(red[0] + b2[0]);
    __syncthreads();
    acc += lmout[(size_t)s * DIM + t] * attn_s;
  }
  agg[(size_t)c * DIM + t] = acc;
}

// clause -> lit: scatter via f32 atomics (edge_src is random).
__global__ void k_edge_c2l(const int* __restrict__ esrc,
                           const float* __restrict__ Mcls,   // clause_h @ ca_ws^T
                           const float* __restrict__ Mlit,   // lit_h @ ca_wd^T
                           const float* __restrict__ b1,
                           const float* __restrict__ w2,
                           const float* __restrict__ b2,
                           const float* __restrict__ cmout,
                           float* __restrict__ agg)
{
  int c = blockIdx.x, t = threadIdx.x;
  __shared__ float red[DIM];
  __shared__ float attn_s;
  float mcls = Mcls[(size_t)c * DIM + t] + b1[t];
  float cm = cmout[(size_t)c * DIM + t];
  for (int j = 0; j < 3; ++j) {
    int l = esrc[c * 3 + j];
    float e = fmaxf(mcls + Mlit[(size_t)l * DIM + t], 0.0f);
    red[t] = e * w2[t];
    __syncthreads();
    for (int off = 64; off > 0; off >>= 1) {
      if (t < off) red[t] += red[t + off];
      __syncthreads();
    }
    if (t == 0) attn_s = sigm(red[0] + b2[0]);
    __syncthreads();
    atomicAdd(&agg[(size_t)l * DIM + t], cm * attn_s);
    __syncthreads();
  }
}

// ---------------------------------------------------------------- LSTM + BN

// g[M,512] gates (i,f,g,o). c updated in place; h <- h + sig(o)*tanh(c_new) (pre-BN).
__global__ void k_lstm_res(const float* __restrict__ g, float* __restrict__ h,
                           float* __restrict__ c, int M)
{
  int i = blockIdx.x * blockDim.x + threadIdx.x;
  if (i >= M * DIM) return;
  int r = i >> 7, t = i & 127;
  const float* gr = g + (size_t)r * GDIM;
  float gi = sigm(gr[t]);
  float gf = sigm(gr[DIM + t]);
  float gg = tanhf(gr[2 * DIM + t]);
  float go = sigm(gr[3 * DIM + t]);
  float cn = gf * c[i] + gi * gg;
  c[i] = cn;
  h[i] = h[i] + go * tanhf(cn);
}

__global__ void k_bn_stats(const float* __restrict__ x, int M,
                           float* __restrict__ mean, float* __restrict__ rstd)
{
  int f = blockIdx.x, t = threadIdx.x;
  float s = 0.0f, s2 = 0.0f;
  for (int r = t; r < M; r += 256) {
    float v = x[(size_t)r * DIM + f];
    s += v; s2 += v * v;
  }
  __shared__ float rs[256], rq[256];
  rs[t] = s; rq[t] = s2;
  __syncthreads();
  for (int off = 128; off > 0; off >>= 1) {
    if (t < off) { rs[t] += rs[t + off]; rq[t] += rq[t + off]; }
    __syncthreads();
  }
  if (t == 0) {
    float m = rs[0] / (float)M;
    float v = rq[0] / (float)M - m * m;
    mean[f] = m;
    rstd[f] = rsqrtf(v + 1e-5f);
  }
}

__global__ void k_bn_apply(float* __restrict__ x, int M,
                           const float* __restrict__ mean, const float* __restrict__ rstd,
                           const float* __restrict__ g, const float* __restrict__ b)
{
  int i = blockIdx.x * blockDim.x + threadIdx.x;
  if (i >= M * DIM) return;
  int f = i & 127;
  x[i] = (x[i] - mean[f]) * rstd[f] * g[f] + b[f];
}

// ---------------------------------------------------------------- vote head

__global__ void k_vote(const float* __restrict__ h2, const float* __restrict__ w3,
                       const float* __restrict__ b3, float* __restrict__ accum)
{
  int v = blockIdx.x, t = threadIdx.x;
  __shared__ float rp[DIM], rn[DIM];
  rp[t] = h2[(size_t)v * DIM + t] * w3[t];
  rn[t] = h2[(size_t)(v + NVARS) * DIM + t] * w3[t];
  __syncthreads();
  for (int off = 64; off > 0; off >>= 1) {
    if (t < off) { rp[t] += rp[t + off]; rn[t] += rn[t + off]; }
    __syncthreads();
  }
  if (t == 0) {
    float pos = rp[0] + b3[0], neg = rn[0] + b3[0];
    float vote = (fabsf(pos) > fabsf(neg)) ? pos : -neg;
    atomicAdd(accum, vote);
  }
}

__global__ void k_finalize(const float* __restrict__ accum, float* __restrict__ out) {
  if (threadIdx.x == 0) out[0] = sigm(accum[0] / (float)NVARS);
}

// ---------------------------------------------------------------- host side

static void gemm(hipStream_t s, const unsigned short* A, int lda,
                 const unsigned short* B, int ldb, float* C, int ldc,
                 const float* bias, int M, int N, int K, int accum, int relu)
{
  int strips = (M / 16) * (N / 64);
  int blocks = (strips + 7) / 8;                 // 256 threads = 8 waves
  gemm_bf16_wmma<<<dim3(blocks), dim3(256), 0, s>>>(A, lda, B, ldb, C, ldc,
                                                    bias, M, N, K, accum, relu);
}

static void conv(hipStream_t s, const float* src, unsigned short* dst, int n) {
  k_f32_to_bf16<<<dim3((n + 255) / 256), dim3(256), 0, s>>>(src, dst, n);
}

static void zero(hipStream_t s, float* p, int n) {
  k_zero<<<dim3((n + 255) / 256), dim3(256), 0, s>>>(p, n);
}

extern "C" void kernel_launch(void* const* d_in, const int* in_sizes, int n_in,
                              void* d_out, int out_size, void* d_ws, size_t ws_size,
                              hipStream_t stream)
{
  (void)in_sizes; (void)n_in; (void)out_size; (void)ws_size;
  const int*   edge_src = (const int*)d_in[0];
  const float* li_w  = (const float*)d_in[4];
  const float* li_b  = (const float*)d_in[5];
  const float* ci_w  = (const float*)d_in[6];
  const float* ci_b  = (const float*)d_in[7];
  const float* la_w1 = (const float*)d_in[8];
  const float* la_b1 = (const float*)d_in[9];
  const float* la_w2 = (const float*)d_in[10];
  const float* la_b2 = (const float*)d_in[11];
  const float* ca_w1 = (const float*)d_in[12];
  const float* ca_b1 = (const float*)d_in[13];
  const float* ca_w2 = (const float*)d_in[14];
  const float* ca_b2 = (const float*)d_in[15];
  const float* lm_w1 = (const float*)d_in[16];
  const float* lm_b1 = (const float*)d_in[17];
  const float* lm_w2 = (const float*)d_in[18];
  const float* lm_b2 = (const float*)d_in[19];
  const float* lm_w3 = (const float*)d_in[20];
  const float* lm_b3 = (const float*)d_in[21];
  const float* cm_w1 = (const float*)d_in[22];
  const float* cm_b1 = (const float*)d_in[23];
  const float* cm_w2 = (const float*)d_in[24];
  const float* cm_b2 = (const float*)d_in[25];
  const float* cm_w3 = (const float*)d_in[26];
  const float* cm_b3 = (const float*)d_in[27];
  const float* ll_wih = (const float*)d_in[28];
  const float* ll_bih = (const float*)d_in[29];
  const float* ll_whh = (const float*)d_in[30];
  const float* ll_bhh = (const float*)d_in[31];
  const float* cl_wih = (const float*)d_in[32];
  const float* cl_bih = (const float*)d_in[33];
  const float* cl_whh = (const float*)d_in[34];
  const float* cl_bhh = (const float*)d_in[35];
  const float* lbn_g = (const float*)d_in[36];
  const float* lbn_b = (const float*)d_in[37];
  const float* cbn_g = (const float*)d_in[38];
  const float* cbn_b = (const float*)d_in[39];
  const float* v_w1  = (const float*)d_in[40];
  const float* v_b1  = (const float*)d_in[41];
  const float* v_w2  = (const float*)d_in[42];
  const float* v_b2  = (const float*)d_in[43];
  const float* v_w3  = (const float*)d_in[44];
  const float* v_b3  = (const float*)d_in[45];

  // ---- workspace carve (256B aligned)
  char* wp = (char*)d_ws;
  auto alloc = [&](size_t bytes) -> void* {
    void* p = (void*)wp;
    wp += (bytes + 255) & ~(size_t)255;
    return p;
  };
  float* lit_h = (float*)alloc((size_t)NLITS * DIM * 4);
  float* lit_c = (float*)alloc((size_t)NLITS * DIM * 4);
  float* cls_h = (float*)alloc((size_t)NCLS * DIM * 4);
  float* cls_c = (float*)alloc((size_t)NCLS * DIM * 4);
  float* P_lit = (float*)alloc((size_t)NLITS * DIM * 4);
  float* P_cls = (float*)alloc((size_t)NCLS * DIM * 4);
  float* mlpA  = (float*)alloc((size_t)NCLS * DIM * 4);
  float* mlpB  = (float*)alloc((size_t)NCLS * DIM * 4);
  float* aggf  = (float*)alloc((size_t)NCLS * DIM * 4);
  float* gbuf  = (float*)alloc((size_t)NCLS * GDIM * 4);
  float* meanb = (float*)alloc(DIM * 4);
  float* rstdb = (float*)alloc(DIM * 4);
  float* accum = (float*)alloc(4);

  unsigned short* lit_h_bf = (unsigned short*)alloc((size_t)NLITS * DIM * 2);
  unsigned short* cls_h_bf = (unsigned short*)alloc((size_t)NCLS * DIM * 2);
  unsigned short* flip_bf  = (unsigned short*)alloc((size_t)NLITS * DIM * 2);
  unsigned short* mlpA_bf  = (unsigned short*)alloc((size_t)NCLS * DIM * 2);
  unsigned short* mlpB_bf  = (unsigned short*)alloc((size_t)NCLS * DIM * 2);
  unsigned short* agg_bf   = (unsigned short*)alloc((size_t)NCLS * DIM * 2);

  unsigned short* la_w1_bf = (unsigned short*)alloc(DIM * 2 * DIM * 2);
  unsigned short* ca_w1_bf = (unsigned short*)alloc(DIM * 2 * DIM * 2);
  unsigned short* lm_w1_bf = (unsigned short*)alloc(DIM * DIM * 2);
  unsigned short* lm_w2_bf = (unsigned short*)alloc(DIM * DIM * 2);
  unsigned short* lm_w3_bf = (unsigned short*)alloc(DIM * DIM * 2);
  unsigned short* cm_w1_bf = (unsigned short*)alloc(DIM * DIM * 2);
  unsigned short* cm_w2_bf = (unsigned short*)alloc(DIM * DIM * 2);
  unsigned short* cm_w3_bf = (unsigned short*)alloc(DIM * DIM * 2);
  unsigned short* ll_wih_bf = (unsigned short*)alloc(GDIM * 2 * DIM * 2);
  unsigned short* ll_whh_bf = (unsigned short*)alloc(GDIM * DIM * 2);
  unsigned short* cl_wih_bf = (unsigned short*)alloc(GDIM * DIM * 2);
  unsigned short* cl_whh_bf = (unsigned short*)alloc(GDIM * DIM * 2);
  unsigned short* v_w1_bf  = (unsigned short*)alloc(DIM * DIM * 2);
  unsigned short* v_w2_bf  = (unsigned short*)alloc(DIM * DIM * 2);

  // ---- weight conversions (deterministic, each call)
  conv(stream, la_w1, la_w1_bf, DIM * 2 * DIM);
  conv(stream, ca_w1, ca_w1_bf, DIM * 2 * DIM);
  conv(stream, lm_w1, lm_w1_bf, DIM * DIM);
  conv(stream, lm_w2, lm_w2_bf, DIM * DIM);
  conv(stream, lm_w3, lm_w3_bf, DIM * DIM);
  conv(stream, cm_w1, cm_w1_bf, DIM * DIM);
  conv(stream, cm_w2, cm_w2_bf, DIM * DIM);
  conv(stream, cm_w3, cm_w3_bf, DIM * DIM);
  conv(stream, ll_wih, ll_wih_bf, GDIM * 2 * DIM);
  conv(stream, ll_whh, ll_whh_bf, GDIM * DIM);
  conv(stream, cl_wih, cl_wih_bf, GDIM * DIM);
  conv(stream, cl_whh, cl_whh_bf, GDIM * DIM);
  conv(stream, v_w1, v_w1_bf, DIM * DIM);
  conv(stream, v_w2, v_w2_bf, DIM * DIM);

  // ---- init state
  k_init_h<<<dim3((NLITS * DIM + 255) / 256), dim3(256), 0, stream>>>(lit_h, li_w, li_b, NLITS * DIM);
  k_init_h<<<dim3((NCLS * DIM + 255) / 256), dim3(256), 0, stream>>>(cls_h, ci_w, ci_b, NCLS * DIM);
  zero(stream, lit_c, NLITS * DIM);
  zero(stream, cls_c, NCLS * DIM);

  // ---- message-passing rounds
  for (int r = 0; r < NROUNDS; ++r) {
    conv(stream, lit_h, lit_h_bf, NLITS * DIM);
    k_flip_to_bf16<<<dim3((NLITS * DIM + 255) / 256), dim3(256), 0, stream>>>(lit_h, flip_bf);
    conv(stream, cls_h, cls_h_bf, NCLS * DIM);

    // attention projections (lit -> clause)
    gemm(stream, lit_h_bf, DIM, la_w1_bf, 2 * DIM, P_lit, DIM, nullptr, NLITS, DIM, DIM, 0, 0);
    gemm(stream, cls_h_bf, DIM, la_w1_bf + DIM, 2 * DIM, P_cls, DIM, nullptr, NCLS, DIM, DIM, 0, 0);

    // lm MLP3 on lit_h
    gemm(stream, lit_h_bf, DIM, lm_w1_bf, DIM, mlpA, DIM, lm_b1, NLITS, DIM, DIM, 0, 1);
    conv(stream, mlpA, mlpA_bf, NLITS * DIM);
    gemm(stream, mlpA_bf, DIM, lm_w2_bf, DIM, mlpB, DIM, lm_b2, NLITS, DIM, DIM, 0, 1);
    conv(stream, mlpB, mlpB_bf, NLITS * DIM);
    gemm(stream, mlpB_bf, DIM, lm_w3_bf, DIM, mlpA, DIM, lm_b3, NLITS, DIM, DIM, 0, 0);

    // edges lit->clause : agg_l
    k_edge_l2c<<<dim3(NCLS), dim3(DIM), 0, stream>>>(edge_src, P_lit, P_cls,
                                                     la_b1, la_w2, la_b2, mlpA, aggf);
    conv(stream, aggf, agg_bf, NCLS * DIM);

    // clause LSTM gates
    gemm(stream, agg_bf, DIM, cl_wih_bf, DIM, gbuf, GDIM, cl_bih, NCLS, GDIM, DIM, 0, 0);
    gemm(stream, cls_h_bf, DIM, cl_whh_bf, DIM, gbuf, GDIM, cl_bhh, NCLS, GDIM, DIM, 1, 0);
    k_lstm_res<<<dim3((NCLS * DIM + 255) / 256), dim3(256), 0, stream>>>(gbuf, cls_h, cls_c, NCLS);
    k_bn_stats<<<dim3(DIM), dim3(256), 0, stream>>>(cls_h, NCLS, meanb, rstdb);
    k_bn_apply<<<dim3((NCLS * DIM + 255) / 256), dim3(256), 0, stream>>>(cls_h, NCLS, meanb, rstdb, cbn_g, cbn_b);
    conv(stream, cls_h, cls_h_bf, NCLS * DIM);

    // attention projections (clause -> lit)
    gemm(stream, cls_h_bf, DIM, ca_w1_bf, 2 * DIM, P_cls, DIM, nullptr, NCLS, DIM, DIM, 0, 0);
    gemm(stream, lit_h_bf, DIM, ca_w1_bf + DIM, 2 * DIM, P_lit, DIM, nullptr, NLITS, DIM, DIM, 0, 0);

    // cm MLP3 on clause_h
    gemm(stream, cls_h_bf, DIM, cm_w1_bf, DIM, mlpA, DIM, cm_b1, NCLS, DIM, DIM, 0, 1);
    conv(stream, mlpA, mlpA_bf, NCLS * DIM);
    gemm(stream, mlpA_bf, DIM, cm_w2_bf, DIM, mlpB, DIM, cm_b2, NCLS, DIM, DIM, 0, 1);
    conv(stream, mlpB, mlpB_bf, NCLS * DIM);
    gemm(stream, mlpB_bf, DIM, cm_w3_bf, DIM, mlpA, DIM, cm_b3, NCLS, DIM, DIM, 0, 0);

    // edges clause->lit : agg_c (scatter, atomics)
    zero(stream, aggf, NLITS * DIM);
    k_edge_c2l<<<dim3(NCLS), dim3(DIM), 0, stream>>>(edge_src, P_cls, P_lit,
                                                     ca_b1, ca_w2, ca_b2, mlpA, aggf);
    conv(stream, aggf, agg_bf, NLITS * DIM);

    // lit LSTM gates: agg_c@Wih[:, :128]^T + flipped@Wih[:,128:]^T + bih + h@Whh^T + bhh
    gemm(stream, agg_bf, DIM, ll_wih_bf, 2 * DIM, gbuf, GDIM, ll_bih, NLITS, GDIM, DIM, 0, 0);
    gemm(stream, flip_bf, DIM, ll_wih_bf + DIM, 2 * DIM, gbuf, GDIM, nullptr, NLITS, GDIM, DIM, 1, 0);
    gemm(stream, lit_h_bf, DIM, ll_whh_bf, DIM, gbuf, GDIM, ll_bhh, NLITS, GDIM, DIM, 1, 0);
    k_lstm_res<<<dim3((NLITS * DIM + 255) / 256), dim3(256), 0, stream>>>(gbuf, lit_h, lit_c, NLITS);
    k_bn_stats<<<dim3(DIM), dim3(256), 0, stream>>>(lit_h, NLITS, meanb, rstdb);
    k_bn_apply<<<dim3((NLITS * DIM + 255) / 256), dim3(256), 0, stream>>>(lit_h, NLITS, meanb, rstdb, lbn_g, lbn_b);
  }

  // ---- vote head
  conv(stream, lit_h, lit_h_bf, NLITS * DIM);
  gemm(stream, lit_h_bf, DIM, v_w1_bf, DIM, mlpA, DIM, v_b1, NLITS, DIM, DIM, 0, 1);
  conv(stream, mlpA, mlpA_bf, NLITS * DIM);
  gemm(stream, mlpA_bf, DIM, v_w2_bf, DIM, mlpB, DIM, v_b2, NLITS, DIM, DIM, 0, 1);
  zero(stream, accum, 1);
  k_vote<<<dim3(NVARS), dim3(DIM), 0, stream>>>(mlpB, v_w3, v_b3, accum);
  k_finalize<<<dim3(1), dim3(32), 0, stream>>>(accum, (float*)d_out);
}
